// ODELSTM_53987738911255
// MI455X (gfx1250) — compile-verified, matrix operand order checked by reading, not run on hardware
//
#include <hip/hip_runtime.h>
#include <hip/hip_bf16.h>
#include <math.h>
#include <stdint.h>

typedef __attribute__((ext_vector_type(16))) __bf16 v16bf;
typedef __attribute__((ext_vector_type(8)))  float  v8f;

#define BB 512   // batch
#define HH 512   // hidden
#define DD 128   // features
#define EE 256   // events per step
#define KK 64    // steps
#define DHC 640  // D + H
#define G4 2048  // 4*H packed gates
#define DT_STEP 0.05f

// ---------- helpers ----------
__device__ __forceinline__ __bf16 f2bf(float f) {
    unsigned u = __float_as_uint(f);
    u += 0x7FFFu + ((u >> 16) & 1u);           // round-to-nearest-even
    unsigned short h = (unsigned short)(u >> 16);
    __bf16 r;
    __builtin_memcpy(&r, &h, 2);
    return r;
}

union FragBF { v16bf v; uint4 q[2]; };

enum { EPI_TANH_BF16 = 0, EPI_EULER = 1, EPI_RELU_BF16 = 2, EPI_F32 = 3 };

// ---------- Tensor Data Mover (gfx1250) ----------
#if __has_builtin(__builtin_amdgcn_tensor_load_to_lds)
#define HAS_TDM 1
#else
#define HAS_TDM 0
#endif

#if HAS_TDM
typedef __attribute__((ext_vector_type(4))) unsigned int u32x4;
typedef __attribute__((ext_vector_type(8))) int          i32x8;
typedef __attribute__((ext_vector_type(4))) int          i32x4;
typedef __attribute__((address_space(3))) char           lds_char_t;

__device__ __forceinline__ unsigned lds_addr_of(const void* p) {
    // generic -> LDS addrspace -> 32-bit LDS byte offset (D#.lds_addr)
    return (unsigned)(uintptr_t)(lds_char_t*)p;
}

// 2D tile DMA: tile_d0 (in 8-byte units) x tile_d1 rows, row stride stride0 (8B units).
// D# group0: count=1 | lds_addr | global_addr[56:0] | type=2.
// D# group1: data_size=3 (8B), tensor dims huge (no OOB clip), tile dims, dim0 stride.
__device__ __forceinline__ void tdm_load_2d(unsigned lds, unsigned long long ga,
                                            unsigned tile_d0, unsigned tile_d1,
                                            unsigned stride0)
{
    u32x4 g0 = { 1u,                                   // count = 1 descriptor
                 lds,                                  // LDS byte address
                 (unsigned)ga,                         // global addr [31:0]
                 (unsigned)((ga >> 32) & 0x01FFFFFFu)  // global addr [56:32]
                 | 0x80000000u };                      // type = 2 ("image")
    const unsigned td0 = 1u << 30, td1 = 1u << 20;     // tensor dims: huge (no clip)
    i32x8 g1 = { (int)(3u << 16),                            // data_size = 3 (8 bytes)
                 (int)((td0 & 0xFFFFu) << 16),               // tensor_dim0[15:0]
                 (int)((td0 >> 16) | ((td1 & 0xFFFFu) << 16)),
                 (int)((td1 >> 16) | (tile_d0 << 16)),       // tile_dim0
                 (int)tile_d1,                               // tile_dim1 (tile_dim2=0)
                 (int)stride0,                               // tensor_dim0_stride[31:0]
                 0, 0 };
    i32x4 z4 = { 0, 0, 0, 0 };
#if defined(__clang_major__) && __clang_major__ >= 23
    i32x8 z8 = { 0, 0, 0, 0, 0, 0, 0, 0 };
    __builtin_amdgcn_tensor_load_to_lds(g0, g1, z4, z4, z8, 0);
#else
    __builtin_amdgcn_tensor_load_to_lds(g0, g1, z4, z4, 0);
#endif
}
#endif // HAS_TDM

// ---------- WMMA GEMM:  out = epilogue(A[M,K] @ Bpacked + bias) ----------
// A bf16 row-major activations. Bp = weights pre-packed as 128x32 LDS tile images:
// element ((nb*(K/32)+kb)*4096 + n*32 + k). TDM double-buffers both tiles into LDS.
// 256 threads = 8 waves; block tile 128x128, BK=32; wave grid 4(M) x 2(N);
// each wave: 2x4 tiles of 16x16 via v_wmma_f32_16x16x32_bf16.
template <int EPI>
__global__ __launch_bounds__(256)
void wmma_gemm_kernel(const __bf16* __restrict__ A, const __bf16* __restrict__ Bp,
                      const float* __restrict__ bias,
                      float* __restrict__ out_f, __bf16* __restrict__ out_b,
                      int Mdim, int Ndim, int Kdim)
{
    constexpr int BM = 128, BN = 128, BK = 32;
    __shared__ __bf16 As[2][BM * BK];   // [m][k] image
    __shared__ __bf16 Bs[2][BN * BK];   // [n][k] image (pre-packed)

    const int tid  = threadIdx.x;
    const int lane = tid & 31;
    const int wave = tid >> 5;
    const int wm = wave & 3;            // 0..3 along M (32 rows each)
    const int wn = wave >> 2;           // 0..1 along N (64 cols each)
    const int m0 = blockIdx.y * BM;
    const int n0 = blockIdx.x * BN;
    const int row  = lane & 15;
    const int half = lane >> 4;
    const int ktiles = Kdim / BK;
    const size_t btile_base = (size_t)blockIdx.x * (size_t)ktiles * (BN * BK);

    v8f acc[2][4];
    #pragma unroll
    for (int i = 0; i < 2; ++i)
        #pragma unroll
        for (int j = 0; j < 4; ++j)
            acc[i][j] = (v8f){0.f,0.f,0.f,0.f,0.f,0.f,0.f,0.f};

#if HAS_TDM
    auto issue = [&](int buf, int kt) {
        if (wave == 0) {
            // A tile: 128 rows x 64B, row stride Kdim*2 bytes
            tdm_load_2d(lds_addr_of(&As[buf][0]),
                        (unsigned long long)(uintptr_t)&A[(size_t)m0 * Kdim + kt * BK],
                        (unsigned)(BK * 2 / 8), (unsigned)BM,
                        (unsigned)(Kdim * 2 / 8));
        } else if (wave == 1) {
            // B tile: one contiguous 8KB pre-packed image
            tdm_load_2d(lds_addr_of(&Bs[buf][0]),
                        (unsigned long long)(uintptr_t)&Bp[btile_base + (size_t)kt * (BN * BK)],
                        (unsigned)(BN * BK * 2 / 8), 1u,
                        (unsigned)(BN * BK * 2 / 8));
        }
    };
    issue(0, 0);
#endif

    for (int kt = 0; kt < ktiles; ++kt) {
        const int cur = kt & 1;
#if HAS_TDM
        const bool more = (kt + 1 < ktiles);
        if (more) issue(cur ^ 1, kt + 1);          // prefetch next tile via TDM
        if (wave < 2) {                            // issuing waves drain their DMA
            if (more) __builtin_amdgcn_s_wait_tensorcnt(1);
            else      __builtin_amdgcn_s_wait_tensorcnt(0);
        }
#else
        for (int i = tid; i < (BM * BK) / 8; i += 256) {
            int r  = i >> 2;
            int ck = (i & 3) << 3;
            *(uint4*)&As[cur][r * BK + ck] =
                *(const uint4*)&A[(size_t)(m0 + r) * Kdim + (kt * BK + ck)];
        }
        for (int i = tid; i < (BN * BK) / 8; i += 256) {
            *(uint4*)&Bs[cur][i * 8] =
                *(const uint4*)&Bp[btile_base + (size_t)kt * (BN * BK) + i * 8];
        }
#endif
        __syncthreads();

        // fragments: ISA 16-bit layout => two contiguous 16B chunks per lane
        FragBF a[2], b[4];
        #pragma unroll
        for (int mi = 0; mi < 2; ++mi) {
            const __bf16* p = &As[cur][(wm * 32 + mi * 16 + row) * BK + half * 8];
            a[mi].q[0] = *(const uint4*)p;
            a[mi].q[1] = *(const uint4*)(p + 16);
        }
        #pragma unroll
        for (int ni = 0; ni < 4; ++ni) {
            const __bf16* p = &Bs[cur][(wn * 64 + ni * 16 + row) * BK + half * 8];
            b[ni].q[0] = *(const uint4*)p;
            b[ni].q[1] = *(const uint4*)(p + 16);
        }
        #pragma unroll
        for (int mi = 0; mi < 2; ++mi)
            #pragma unroll
            for (int ni = 0; ni < 4; ++ni)
                acc[mi][ni] = __builtin_amdgcn_wmma_f32_16x16x32_bf16(
                    false, a[mi].v, false, b[ni].v,
                    (short)0, acc[mi][ni], false, false);
        __syncthreads();   // protect buffers before next DMA overwrites
    }

    // epilogue: D layout = VGPR r -> row (r + 8*half), col = lane&15
    const int col = lane & 15;
    #pragma unroll
    for (int mi = 0; mi < 2; ++mi) {
        #pragma unroll
        for (int ni = 0; ni < 4; ++ni) {
            const int gc = n0 + wn * 64 + ni * 16 + col;
            const float bv = bias[gc];
            #pragma unroll
            for (int r = 0; r < 8; ++r) {
                const int gr = m0 + wm * 32 + mi * 16 + half * 8 + r;
                const size_t o = (size_t)gr * Ndim + gc;
                float v = acc[mi][ni][r] + bv;
                if (EPI == EPI_TANH_BF16) {
                    out_b[o] = f2bf(tanhf(v));
                } else if (EPI == EPI_RELU_BF16) {
                    out_b[o] = f2bf(fmaxf(v, 0.f));
                } else if (EPI == EPI_F32) {
                    out_f[o] = v;
                } else { // EPI_EULER: h += dt*v ; f32 master + bf16 shadow
                    float hn = out_f[o] + DT_STEP * v;
                    out_f[o] = hn;
                    out_b[o] = f2bf(hn);
                }
            }
        }
    }
}

// ---------- init / weight packing ----------
__global__ __launch_bounds__(256)
void zero_init_kernel(float* __restrict__ h, float* __restrict__ c,
                      __bf16* __restrict__ hb, float* __restrict__ scal)
{
    for (int i = blockIdx.x * blockDim.x + threadIdx.x; i < BB * HH;
         i += gridDim.x * blockDim.x) {
        h[i] = 0.f; c[i] = 0.f; hb[i] = f2bf(0.f);
    }
    if (blockIdx.x == 0 && threadIdx.x == 0) { scal[0] = 0.f; scal[1] = 0.f; }
}

// f32 [K,N] row-major -> bf16 packed tile images [(nb*(K/32)+kb)*4096 + n*32 + k]
__global__ __launch_bounds__(256)
void pack_w_tiles_kernel(const float* __restrict__ W, __bf16* __restrict__ dst,
                         int K, int N)
{
    int i = blockIdx.x * 256 + threadIdx.x;
    if (i >= K * N) return;
    int tile   = i >> 12;
    int within = i & 4095;
    int n  = within >> 5;
    int kk = within & 31;
    int kt = K >> 5;
    int kb = tile % kt;
    int nb = tile / kt;
    dst[i] = f2bf(W[(size_t)(kb * 32 + kk) * N + (nb * 128 + n)]);
}

// 4 gate weights [640,512] -> packed [640,2048] tile images + packed bias
__global__ __launch_bounds__(256)
void pack_g_tiles_kernel(const float* __restrict__ Wi, const float* __restrict__ Wf,
                         const float* __restrict__ Wo, const float* __restrict__ Wc,
                         const float* __restrict__ bi, const float* __restrict__ bfv,
                         const float* __restrict__ bo, const float* __restrict__ bc,
                         __bf16* __restrict__ dst, float* __restrict__ bg)
{
    int i = blockIdx.x * 256 + threadIdx.x;        // DHC*G4
    int tile   = i >> 12;
    int within = i & 4095;
    int n  = within >> 5;
    int kk = within & 31;
    int kt = DHC >> 5;                             // 20
    int kb = tile % kt;
    int nb = tile / kt;
    int r = kb * 32 + kk;
    int c = nb * 128 + n;
    int g = c >> 9, cc = c & 511;
    const float* Ws = (g == 0) ? Wi : (g == 1) ? Wf : (g == 2) ? Wo : Wc;
    dst[i] = f2bf(Ws[(size_t)r * 512 + cc]);
    if (i < G4) {
        int gg = i >> 9, c2 = i & 511;
        const float* bs = (gg == 0) ? bi : (gg == 1) ? bfv : (gg == 2) ? bo : bc;
        bg[i] = bs[c2];
    }
}

__global__ __launch_bounds__(256)
void totm_kernel(const float* __restrict__ Mv, float* __restrict__ acc, int n)
{
    float s = 0.f;
    for (int i = blockIdx.x * 256 + threadIdx.x; i < n; i += gridDim.x * 256)
        s += Mv[i];
    __shared__ float red[256];
    red[threadIdx.x] = s; __syncthreads();
    for (int st = 128; st > 0; st >>= 1) {
        if (threadIdx.x < st) red[threadIdx.x] += red[threadIdx.x + st];
        __syncthreads();
    }
    if (threadIdx.x == 0) atomicAdd(acc, red[0]);
}

// ---------- per-event: loss + gather + build LSTM input ----------
__global__ __launch_bounds__(128)
void loss_gather_kernel(const float* __restrict__ X, const float* __restrict__ Mv,
                        const int* __restrict__ bidx, const float* __restrict__ p,
                        const __bf16* __restrict__ h_bf, __bf16* __restrict__ inp_bf,
                        float* __restrict__ loss_acc, int k)
{
    const int e = blockIdx.x;
    const int t = threadIdx.x;                     // 0..127 (== DD)
    const int idx = bidx[k * EE + e];
    const size_t eo = (size_t)k * EE + e;
    float xo = X[eo * DD + t];
    float part = fabsf(xo - p[(size_t)idx * DD + t]) * Mv[eo * DD + t];
    inp_bf[(size_t)e * DHC + t] = f2bf(xo);
    for (int j = t; j < HH; j += 128)
        inp_bf[(size_t)e * DHC + DD + j] = h_bf[(size_t)idx * HH + j];
    __shared__ float red[128];
    red[t] = part; __syncthreads();
    for (int s = 64; s > 0; s >>= 1) {
        if (t < s) red[t] += red[t + s];
        __syncthreads();
    }
    if (t == 0) atomicAdd(loss_acc, red[0]);
}

// ---------- per-event: LSTM pointwise update, scatter into h/c ----------
__global__ __launch_bounds__(256)
void lstm_update_kernel(const float* __restrict__ gates, const int* __restrict__ bidx,
                        float* __restrict__ c_f, float* __restrict__ h_f,
                        __bf16* __restrict__ h_b, int k)
{
    int i = blockIdx.x * 256 + threadIdx.x;        // EE*HH
    int e = i >> 9;
    int j = i & 511;
    int idx = bidx[k * EE + e];
    const float* g = gates + (size_t)e * G4;
    float ig = 1.f / (1.f + __expf(-g[j]));
    float fg = 1.f / (1.f + __expf(-g[512 + j]));
    float og = 1.f / (1.f + __expf(-g[1024 + j]));
    float ct = tanhf(g[1536 + j]);
    size_t o = (size_t)idx * HH + j;
    float cn = fg * c_f[o] + ig * ct;
    float hn = og * tanhf(cn);
    c_f[o] = cn; h_f[o] = hn; h_b[o] = f2bf(hn);
}

__global__ void finalize_kernel(const float* __restrict__ scal, float* __restrict__ out)
{
    out[0] = scal[0];
    out[1] = scal[0] / scal[1];
}

// ---------- host ----------
extern "C" void kernel_launch(void* const* d_in, const int* in_sizes, int n_in,
                              void* d_out, int out_size, void* d_ws, size_t ws_size,
                              hipStream_t stream)
{
    const float* X    = (const float*)d_in[0];
    const float* Mv   = (const float*)d_in[1];
    const int*   bidx = (const int*)d_in[2];
    // d_in[3] = sample_idx (identity, unused)
    const float* Wi  = (const float*)d_in[4];  const float* bi  = (const float*)d_in[5];
    const float* Wf  = (const float*)d_in[6];  const float* bfv = (const float*)d_in[7];
    const float* Wo  = (const float*)d_in[8];  const float* bo  = (const float*)d_in[9];
    const float* Wc  = (const float*)d_in[10]; const float* bc  = (const float*)d_in[11];
    const float* W1  = (const float*)d_in[12]; const float* b1  = (const float*)d_in[13];
    const float* W2  = (const float*)d_in[14]; const float* b2  = (const float*)d_in[15];
    const float* Wp1 = (const float*)d_in[16]; const float* bp1 = (const float*)d_in[17];
    const float* Wp2 = (const float*)d_in[18]; const float* bp2 = (const float*)d_in[19];

    char* ws = (char*)d_ws;
    size_t off = 0;
    auto alloc = [&](size_t bytes) { size_t o = off; off = (off + bytes + 511) & ~(size_t)511; return o; };

    float*  h_f   = (float*) (ws + alloc((size_t)BB * HH * 4));
    float*  c_f   = (float*) (ws + alloc((size_t)BB * HH * 4));
    __bf16* h_b   = (__bf16*)(ws + alloc((size_t)BB * HH * 2));
    __bf16* t_b   = (__bf16*)(ws + alloc((size_t)BB * HH * 2));
    __bf16* z_b   = (__bf16*)(ws + alloc((size_t)BB * HH * 2));
    float*  p_f   = (float*) (ws + alloc((size_t)BB * DD * 4));
    __bf16* inp_b = (__bf16*)(ws + alloc((size_t)EE * DHC * 2));
    float*  gates = (float*) (ws + alloc((size_t)EE * G4 * 4));
    __bf16* W1p   = (__bf16*)(ws + alloc((size_t)HH * HH * 2));
    __bf16* W2p   = (__bf16*)(ws + alloc((size_t)HH * HH * 2));
    __bf16* Wp1p  = (__bf16*)(ws + alloc((size_t)HH * HH * 2));
    __bf16* Wp2p  = (__bf16*)(ws + alloc((size_t)HH * DD * 2));
    __bf16* Wgp   = (__bf16*)(ws + alloc((size_t)DHC * G4 * 2));
    float*  bg    = (float*) (ws + alloc((size_t)G4 * 4));
    float*  scal  = (float*) (ws + alloc(2 * 4));   // [loss, tot_m]

    // --- prep: state zero + pack weights into WMMA tile images (runs every call) ---
    zero_init_kernel<<<1024, 256, 0, stream>>>(h_f, c_f, h_b, scal);
    pack_w_tiles_kernel<<<(HH * HH) / 256, 256, 0, stream>>>(W1,  W1p,  HH, HH);
    pack_w_tiles_kernel<<<(HH * HH) / 256, 256, 0, stream>>>(W2,  W2p,  HH, HH);
    pack_w_tiles_kernel<<<(HH * HH) / 256, 256, 0, stream>>>(Wp1, Wp1p, HH, HH);
    pack_w_tiles_kernel<<<(HH * DD) / 256, 256, 0, stream>>>(Wp2, Wp2p, HH, DD);
    pack_g_tiles_kernel<<<(DHC * G4) / 256, 256, 0, stream>>>(
        Wi, Wf, Wo, Wc, bi, bfv, bo, bc, Wgp, bg);
    totm_kernel<<<2048, 256, 0, stream>>>(Mv, scal + 1, KK * EE * DD);

    // --- sequential scan over events ---
    for (int k = 0; k < KK; ++k) {
        for (int s = 0; s < 2; ++s) {  // S_EULER substeps
            wmma_gemm_kernel<EPI_TANH_BF16><<<dim3(4, 4), 256, 0, stream>>>(
                h_b, W1p, b1, (float*)nullptr, t_b, BB, HH, HH);
            wmma_gemm_kernel<EPI_EULER><<<dim3(4, 4), 256, 0, stream>>>(
                t_b, W2p, b2, h_f, h_b, BB, HH, HH);
        }
        wmma_gemm_kernel<EPI_RELU_BF16><<<dim3(4, 4), 256, 0, stream>>>(
            h_b, Wp1p, bp1, (float*)nullptr, z_b, BB, HH, HH);
        wmma_gemm_kernel<EPI_F32><<<dim3(1, 4), 256, 0, stream>>>(
            z_b, Wp2p, bp2, p_f, (__bf16*)nullptr, BB, DD, HH);
        loss_gather_kernel<<<EE, 128, 0, stream>>>(X, Mv, bidx, p_f, h_b, inp_b, scal, k);
        wmma_gemm_kernel<EPI_F32><<<dim3(G4 / 128, EE / 128), 256, 0, stream>>>(
            inp_b, Wgp, bg, gates, (__bf16*)nullptr, EE, G4, DHC);
        lstm_update_kernel<<<(EE * HH) / 256, 256, 0, stream>>>(gates, bidx, c_f, h_f, h_b, k);
    }
    // prop_to_end Euler steps don't influence [loss, loss/tot_m] -> skipped.

    finalize_kernel<<<1, 1, 0, stream>>>(scal, (float*)d_out);
}
